// CausalSelfAttention_14654428414506
// MI455X (gfx1250) — matrix operand order repaired and checked by reading, not compiled, not run on hardware
//
#include <hip/hip_runtime.h>
#include <hip/hip_bf16.h>

// ---------------------------------------------------------------------------
// CDNA5 (gfx1250) causal MQA attention block, bf16 WMMA + fp32 softmax path.
// GEMMs stage tiles through LDS via GLOBAL_LOAD_ASYNC_TO_LDS (ASYNCcnt).
// ---------------------------------------------------------------------------

typedef __bf16        v16bf __attribute__((ext_vector_type(16)));
typedef float         v8f   __attribute__((ext_vector_type(8)));
typedef unsigned int  u32x4 __attribute__((ext_vector_type(4)));
typedef int           v4i   __attribute__((ext_vector_type(4)));

static constexpr int Bc = 4;
static constexpr int Tc = 2048;
static constexpr int Ec = 768;
static constexpr int Hc = 6;
static constexpr int Dc = 128;
static constexpr int BT = Bc * Tc;

// ---- gfx1250 async global->LDS path (guarded; fallback = load + ds_store) ----
#if defined(__has_builtin)
# if __has_builtin(__builtin_amdgcn_global_load_async_to_lds_b128) && \
     __has_builtin(__builtin_amdgcn_s_wait_asynccnt)
#  define HAVE_ASYNC_LDS 1
# endif
#endif
#ifndef HAVE_ASYNC_LDS
# define HAVE_ASYNC_LDS 0
#endif

// builtin wants: (AS1 int4*, AS3 int4*, imm offset, imm cpol)
#define GAS1(p) ((__attribute__((address_space(1))) v4i*)(p))
#define LAS3(p) ((__attribute__((address_space(3))) v4i*)(p))

__device__ __forceinline__ unsigned short f2bf(float f) {
    unsigned int u = __builtin_bit_cast(unsigned int, f);
    unsigned int r = u + 0x7FFFu + ((u >> 16) & 1u);   // round-to-nearest-even
    return (unsigned short)(r >> 16);
}

// ---------------------------------------------------------------------------
// fp32 -> bf16 bulk convert
// ---------------------------------------------------------------------------
__global__ void cvt_f32_bf16(const float* __restrict__ s,
                             unsigned short* __restrict__ d, int n) {
    int i = blockIdx.x * 256 + threadIdx.x;
    if (i < n) d[i] = f2bf(s[i]);
}

// ---------------------------------------------------------------------------
// C(M,N) = A(M,K) @ W(N,K)^T   A,W bf16, C fp32.
// Block: 256 threads (8 waves), 64x64 C tile, K-step 32.
// A/B tiles staged to LDS with async global->LDS; each wave computes two
// 16x16 WMMA subtiles (A-fragment reused across both).
// M,N multiples of 64; K multiple of 32.
// ---------------------------------------------------------------------------
__global__ void gemm_bf16_nt(const unsigned short* __restrict__ A,
                             const unsigned short* __restrict__ W,
                             float* __restrict__ C, int M, int N, int K) {
    constexpr int BM = 64, BN = 64, BK = 32;
    __shared__ unsigned short As[BM * BK];   // 64x32 bf16, row = m-local
    __shared__ unsigned short Bs[BN * BK];   // 64x32 bf16, row = n-local

    const int tid  = threadIdx.x;
    const int lane = tid & 31;
    const int w    = tid >> 5;
    const int r    = lane & 15;
    const int hh   = lane >> 4;

    const int nb = N / BN;
    const int m0 = (blockIdx.x / nb) * BM;
    const int n0 = (blockIdx.x % nb) * BN;

    // staging: 256 threads x 16B = full 4KB tile each for A and B
    const int srow = tid >> 2;            // 0..63
    const int sq   = (tid & 3) * 8;       // 8 bf16 = 16B slice
    const unsigned short* gA = A + (size_t)(m0 + srow) * K + sq;
    const unsigned short* gB = W + (size_t)(n0 + srow) * K + sq;
    unsigned short* lA = &As[srow * BK + sq];
    unsigned short* lB = &Bs[srow * BK + sq];

    const int rb  = w & 3;                // row block 0..3
    const int cb0 = (w >> 2) * 2;         // two adjacent col blocks

    v8f acc0 = {0.f, 0.f, 0.f, 0.f, 0.f, 0.f, 0.f, 0.f};
    v8f acc1 = {0.f, 0.f, 0.f, 0.f, 0.f, 0.f, 0.f, 0.f};

    for (int k0 = 0; k0 < K; k0 += BK) {
        __syncthreads();                  // previous compute done before overwrite
#if HAVE_ASYNC_LDS
        __builtin_amdgcn_global_load_async_to_lds_b128(GAS1(gA + k0), LAS3(lA), 0, 0);
        __builtin_amdgcn_global_load_async_to_lds_b128(GAS1(gB + k0), LAS3(lB), 0, 0);
        __builtin_amdgcn_s_wait_asynccnt(0);
#else
        *(u32x4*)lA = *(const u32x4*)(gA + k0);
        *(u32x4*)lB = *(const u32x4*)(gB + k0);
#endif
        if (k0 + BK < K) {
            __builtin_prefetch(gA + k0 + BK, 0, 1);
            __builtin_prefetch(gB + k0 + BK, 0, 1);
        }
        __syncthreads();                  // tile visible block-wide

        // A fragment: lane r = row; half hh holds K 0..15 / 16..31
        union { u32x4 u[2]; v16bf v; } af, b0, b1;
        const unsigned short* ar = &As[(rb * 16 + r) * BK];
        af.u[0] = *(const u32x4*)(ar + 16 * hh);
        af.u[1] = *(const u32x4*)(ar + 16 * hh + 8);
        // B fragments: lanes 0-15 hold K 0..15, lanes 16-31 hold K 16..31, col = r
        const unsigned short* br0 = &Bs[(cb0 * 16 + r) * BK + 16 * hh];
        b0.u[0] = *(const u32x4*)(br0);
        b0.u[1] = *(const u32x4*)(br0 + 8);
        const unsigned short* br1 = &Bs[((cb0 + 1) * 16 + r) * BK + 16 * hh];
        b1.u[0] = *(const u32x4*)(br1);
        b1.u[1] = *(const u32x4*)(br1 + 8);

        acc0 = __builtin_amdgcn_wmma_f32_16x16x32_bf16(false, af.v, false, b0.v,
                                                       (short)0, acc0, false, false);
        acc1 = __builtin_amdgcn_wmma_f32_16x16x32_bf16(false, af.v, false, b1.v,
                                                       (short)0, acc1, false, false);
    }

#pragma unroll
    for (int i = 0; i < 8; i++) {
        const size_t row = (size_t)(m0 + rb * 16 + i + 8 * hh);
        C[row * N + n0 + cb0 * 16 + r]        = acc0[i];
        C[row * N + n0 + (cb0 + 1) * 16 + r]  = acc1[i];
    }
}

// ---------------------------------------------------------------------------
// Per-(b,t): rms_norm + RoPE on q,k (fp32), lambda-mix v, gate; emit bf16
// q (B,H,T,D), k (B,T,D), vT (B,D,T), gate fp32 (B,T,H).
// ---------------------------------------------------------------------------
__global__ void postproc(const float* __restrict__ qf, const float* __restrict__ kf,
                         const float* __restrict__ vf, const float* __restrict__ v1,
                         const float* __restrict__ x,  const float* __restrict__ Wg,
                         const float* __restrict__ lambp,
                         unsigned short* __restrict__ qb, unsigned short* __restrict__ kb,
                         unsigned short* __restrict__ vbT, float* __restrict__ gate) {
    const int bt = blockIdx.x;            // b*T + t
    const int b  = bt / Tc;
    const int t  = bt % Tc;
    const int d  = threadIdx.x;           // 0..127
    const float lamb = *lambp;

    __shared__ float red[Dc];
    __shared__ float nrm[Dc];

    float c = 0.f, s_ = 0.f;
    if (d < 64) {
        float inv = __powf(10000.f, -(float)d / 64.f);
        float fr  = (float)t * inv;
        c  = __cosf(fr);
        s_ = __sinf(fr);
    }

    // ---- K: rms_norm + rope ----
    float kv = kf[(size_t)bt * Dc + d];
    red[d] = kv * kv;
    __syncthreads();
    for (int st = 64; st > 0; st >>= 1) { if (d < st) red[d] += red[d + st]; __syncthreads(); }
    float rn = rsqrtf(red[0] / (float)Dc + 1e-6f);
    nrm[d] = kv * rn;
    __syncthreads();
    if (d < 64) {
        float o0 =  nrm[d] * c  + nrm[d + 64] * s_;
        float o1 = -nrm[d] * s_ + nrm[d + 64] * c;
        kb[(size_t)bt * Dc + d]      = f2bf(o0);
        kb[(size_t)bt * Dc + d + 64] = f2bf(o1);
    }

    // ---- V: lambda mix, store transposed (b, D, T) ----
    float vv = (1.f - lamb) * vf[(size_t)bt * Dc + d] + lamb * v1[(size_t)bt * Dc + d];
    vbT[((size_t)b * Dc + d) * Tc + t] = f2bf(vv);

    // ---- Q per head ----
    for (int h = 0; h < Hc; h++) {
        __syncthreads();
        float qv = qf[(size_t)bt * Ec + h * Dc + d];
        red[d] = qv * qv;
        __syncthreads();
        for (int st = 64; st > 0; st >>= 1) { if (d < st) red[d] += red[d + st]; __syncthreads(); }
        float rq = rsqrtf(red[0] / (float)Dc + 1e-6f);
        nrm[d] = qv * rq;
        __syncthreads();
        if (d < 64) {
            float o0 =  nrm[d] * c  + nrm[d + 64] * s_;
            float o1 = -nrm[d] * s_ + nrm[d + 64] * c;
            size_t base = (((size_t)(b * Hc + h)) * Tc + t) * Dc;
            qb[base + d]      = f2bf(o0);
            qb[base + d + 64] = f2bf(o1);
        }
    }

    // ---- gate = sigmoid(x[:,:12] @ Wg^T) ----
    if (d < Hc) {
        float g = 0.f;
        for (int j = 0; j < 12; j++) g += x[(size_t)bt * Ec + j] * Wg[d * 12 + j];
        gate[(size_t)bt * Hc + d] = 1.f / (1.f + __expf(-g));
    }
}

// ---------------------------------------------------------------------------
// Flash-style causal attention, one wave per (b, h, 16 query rows).
// S = Q K^T * 0.1 (bf16 WMMA, fp32 acc), online softmax in fp32,
// O += P V (bf16 WMMA), epilogue folds gate / row-sum, writes bf16 y.
// ---------------------------------------------------------------------------
__global__ void attn(const unsigned short* __restrict__ qb,
                     const unsigned short* __restrict__ kb,
                     const unsigned short* __restrict__ vbT,
                     const float* __restrict__ gate,
                     unsigned short* __restrict__ yb) {
    __shared__ float pbuf[8][16][32];      // per-wave private P staging

    const int lane = threadIdx.x & 31;
    const int w    = threadIdx.x >> 5;
    const int r    = lane & 15;
    const int hh   = lane >> 4;

    const int task = blockIdx.x * 8 + w;   // b*H*(T/16) tasks
    if (task >= Bc * Hc * (Tc / 16)) return;
    const int qt = task % (Tc / 16);
    const int bh = task / (Tc / 16);
    const int h  = bh % Hc;
    const int b  = bh / Hc;
    const int q0 = qt * 16;

    // Q fragments: 16 rows x 128 head-dim = 4 A-frags
    v16bf qfrag[4];
    const unsigned short* qrow = qb + (((size_t)(b * Hc + h)) * Tc + q0 + r) * Dc;
#pragma unroll
    for (int dj = 0; dj < 4; dj++) {
        union { u32x4 u[2]; v16bf v; } pk;
        pk.u[0] = *(const u32x4*)(qrow + dj * 32 + 16 * hh);
        pk.u[1] = *(const u32x4*)(qrow + dj * 32 + 16 * hh + 8);
        qfrag[dj] = pk.v;
    }

    v8f o[8];
    float m[8], l[8];
#pragma unroll
    for (int i = 0; i < 8; i++) {
        o[i] = (v8f){0.f, 0.f, 0.f, 0.f, 0.f, 0.f, 0.f, 0.f};
        m[i] = -1e30f;
        l[i] = 0.f;
    }

    const int nch = ((q0 + 15) >> 5) + 1;          // causal chunk bound (32 keys/chunk)
    for (int kc = 0; kc < nch; kc++) {
        const int kbase = kc * 32;

        // ---- S = Q K^T over two 16-key column blocks ----
        v8f sfr[2];
#pragma unroll
        for (int cblk = 0; cblk < 2; cblk++) {
            v8f s = {0.f, 0.f, 0.f, 0.f, 0.f, 0.f, 0.f, 0.f};
            const unsigned short* krow =
                kb + ((size_t)b * Tc + kbase + cblk * 16 + r) * Dc + 16 * hh;
#pragma unroll
            for (int dj = 0; dj < 4; dj++) {
                union { u32x4 u[2]; v16bf v; } pk;
                pk.u[0] = *(const u32x4*)(krow + dj * 32);
                pk.u[1] = *(const u32x4*)(krow + dj * 32 + 8);
                s = __builtin_amdgcn_wmma_f32_16x16x32_bf16(false, qfrag[dj], false, pk.v,
                                                            (short)0, s, false, false);
            }
            const int key = kbase + cblk * 16 + r;
#pragma unroll
            for (int i = 0; i < 8; i++) {
                const int qg = q0 + i + 8 * hh;
                s[i] = (key <= qg) ? s[i] * 0.1f : -1e30f;
            }
            sfr[cblk] = s;
        }

        // ---- online softmax update ----
#pragma unroll
        for (int i = 0; i < 8; i++) {
            float mx = fmaxf(sfr[0][i], sfr[1][i]);
            mx = fmaxf(mx, __shfl_xor(mx, 1, 32));
            mx = fmaxf(mx, __shfl_xor(mx, 2, 32));
            mx = fmaxf(mx, __shfl_xor(mx, 4, 32));
            mx = fmaxf(mx, __shfl_xor(mx, 8, 32));   // xor<16 stays in 16-lane half
            const float mn    = fmaxf(m[i], mx);
            const float alpha = __expf(m[i] - mn);
            const float p0 = __expf(sfr[0][i] - mn);
            const float p1 = __expf(sfr[1][i] - mn);
            pbuf[w][i + 8 * hh][r]      = p0;        // C-layout -> LDS (same-wave order)
            pbuf[w][i + 8 * hh][16 + r] = p1;
            float ps = p0 + p1;
            ps += __shfl_xor(ps, 1, 32);
            ps += __shfl_xor(ps, 2, 32);
            ps += __shfl_xor(ps, 4, 32);
            ps += __shfl_xor(ps, 8, 32);
            l[i] = l[i] * alpha + ps;
            m[i] = mn;
#pragma unroll
            for (int dc = 0; dc < 8; dc++) o[dc][i] *= alpha;
        }

        // ---- P as A-fragment (transpose via per-wave LDS) ----
        union { unsigned short s[16]; v16bf v; } pf;
#pragma unroll
        for (int e = 0; e < 16; e++) {
            const int col = 16 * hh + e;
            pf.s[e] = f2bf(pbuf[w][r][col]);
        }

        // ---- O += P V ----
#pragma unroll
        for (int dc = 0; dc < 8; dc++) {
            const unsigned short* vrow =
                vbT + ((size_t)b * Dc + dc * 16 + r) * Tc + kbase + 16 * hh;
            union { u32x4 u[2]; v16bf v; } vk;
            vk.u[0] = *(const u32x4*)(vrow);
            vk.u[1] = *(const u32x4*)(vrow + 8);
            o[dc] = __builtin_amdgcn_wmma_f32_16x16x32_bf16(false, pf.v, false, vk.v,
                                                            (short)0, o[dc], false, false);
        }
    }

    // ---- epilogue: y = gate * O / l ----
    float gsc[8];
#pragma unroll
    for (int i = 0; i < 8; i++) {
        const int row = q0 + i + 8 * hh;
        gsc[i] = gate[((size_t)b * Tc + row) * Hc + h] / l[i];
    }
#pragma unroll
    for (int dc = 0; dc < 8; dc++) {
#pragma unroll
        for (int i = 0; i < 8; i++) {
            const int row = q0 + i + 8 * hh;
            yb[((size_t)b * Tc + row) * Ec + h * Dc + dc * 16 + r] =
                f2bf(o[dc][i] * gsc[i]);
        }
    }
}

// ---------------------------------------------------------------------------
// Host launcher
// ---------------------------------------------------------------------------
extern "C" void kernel_launch(void* const* d_in, const int* in_sizes, int n_in,
                              void* d_out, int out_size, void* d_ws, size_t ws_size,
                              hipStream_t stream) {
    (void)in_sizes; (void)n_in; (void)out_size; (void)ws_size;
    const float* x    = (const float*)d_in[0];
    const float* v1   = (const float*)d_in[1];
    const float* Wq   = (const float*)d_in[2];
    const float* Wk   = (const float*)d_in[3];
    const float* Wv   = (const float*)d_in[4];
    const float* lamb = (const float*)d_in[5];
    const float* Wo   = (const float*)d_in[6];
    const float* Wg   = (const float*)d_in[7];
    float* out = (float*)d_out;

    char* ws = (char*)d_ws;
    size_t off = 0;
    auto alloc = [&](size_t bytes) -> char* {
        char* p = ws + off;
        off = (off + bytes + 255) & ~(size_t)255;
        return p;
    };
    unsigned short* xb  = (unsigned short*)alloc((size_t)BT * Ec * 2);
    unsigned short* wqb = (unsigned short*)alloc((size_t)Ec * Ec * 2);
    unsigned short* wkb = (unsigned short*)alloc((size_t)Dc * Ec * 2);
    unsigned short* wvb = (unsigned short*)alloc((size_t)Dc * Ec * 2);
    unsigned short* wob = (unsigned short*)alloc((size_t)Ec * Ec * 2);
    float*          qf  = (float*)alloc((size_t)BT * Ec * 4);
    float*          kf  = (float*)alloc((size_t)BT * Dc * 4);
    float*          vf  = (float*)alloc((size_t)BT * Dc * 4);
    unsigned short* qb  = (unsigned short*)alloc((size_t)BT * Ec * 2);
    unsigned short* kb  = (unsigned short*)alloc((size_t)BT * Dc * 2);
    unsigned short* vbT = (unsigned short*)alloc((size_t)BT * Dc * 2);
    float*          gt  = (float*)alloc((size_t)BT * Hc * 4);
    unsigned short* yb  = (unsigned short*)alloc((size_t)BT * Ec * 2);

    auto cvt = [&](const float* s, unsigned short* d, int n) {
        cvt_f32_bf16<<<(n + 255) / 256, 256, 0, stream>>>(s, d, n);
    };
    cvt(x,  xb,  BT * Ec);
    cvt(Wq, wqb, Ec * Ec);
    cvt(Wk, wkb, Dc * Ec);
    cvt(Wv, wvb, Dc * Ec);
    cvt(Wo, wob, Ec * Ec);

    auto gemm = [&](const unsigned short* A, const unsigned short* W, float* C,
                    int M, int N, int K) {
        int blocks = (M / 64) * (N / 64);
        gemm_bf16_nt<<<blocks, 256, 0, stream>>>(A, W, C, M, N, K);
    };
    gemm(xb, wqb, qf, BT, Ec, Ec);   // Q projection
    gemm(xb, wkb, kf, BT, Dc, Ec);   // K projection
    gemm(xb, wvb, vf, BT, Dc, Ec);   // V projection

    postproc<<<BT, Dc, 0, stream>>>(qf, kf, vf, v1, x, Wg, lamb, qb, kb, vbT, gt);

    {
        int tasks = Bc * Hc * (Tc / 16);
        attn<<<(tasks + 7) / 8, 256, 0, stream>>>(qb, kb, vbT, gt, yb);
    }

    gemm(yb, wob, out, BT, Ec, Ec);  // output projection -> d_out[0 : B*T*E]

    // second tuple element: v1 passthrough
    (void)hipMemcpyAsync(out + (size_t)BT * Ec, v1, (size_t)BT * Dc * sizeof(float),
                         hipMemcpyDeviceToDevice, stream);
}
